// EGNNConv_1829656068677
// MI455X (gfx1250) — matrix operand order repaired and checked by reading, not compiled, not run on hardware
//
#include <hip/hip_runtime.h>
#include <hip/hip_bf16.h>

typedef __attribute__((ext_vector_type(16))) _Float16 v16h;
typedef __attribute__((ext_vector_type(8)))  float    v8f;

union FragU {
    v16h   h;
    float4 f4[2];
};

#define NN 40000
#define EE 640000
#define DD 128
#define HH 128

// workspace layout (bytes), all 256-aligned
#define OFF_H16 0                        // N*128 f16  = 10,240,000
#define OFF_W1P 10240000                 // 8*8*1024   =     65,536
#define OFF_W2P 10305536                 // 8*4*1024   =     32,768
#define OFF_W3P 10338304                 // 8*8*1024   =     65,536
#define OFF_W4P 10403840                 // 8*4*1024   =     32,768
#define OFF_AGG 10436608                 // N*128 f32  = 20,480,000
// total ~30.9 MB

__device__ __forceinline__ float silu_f(float x) {
    return x * __builtin_amdgcn_rcpf(1.0f + __expf(-x));
}
__device__ __forceinline__ float sigmoid_f(float x) {
    return __builtin_amdgcn_rcpf(1.0f + __expf(-x));
}

// async global -> LDS copy of 16 bytes (gfx1250 GLOBAL_LOAD_ASYNC_TO_LDS_B128).
// lds_off is the wave-relative LDS byte offset (low 32 bits of a generic
// pointer to a __shared__ object, per ISA 10.2 aperture mapping).
__device__ __forceinline__ void async_copy_b128(unsigned lds_off, const void* gaddr) {
    asm volatile("global_load_async_to_lds_b128 %0, %1, off"
                 :: "v"(lds_off), "v"(gaddr) : "memory");
}
__device__ __forceinline__ void wait_asynccnt0() {
    asm volatile("s_wait_asynccnt 0x0" ::: "memory");
}

// ---------------- zero agg ----------------
__global__ void k_zero_f4(float4* p, int n4) {
    int i = blockIdx.x * blockDim.x + threadIdx.x;
    if (i < n4) p[i] = make_float4(0.f, 0.f, 0.f, 0.f);
}

// ---------------- f32 -> f16 convert ----------------
__global__ void k_cvt_h16(const float* __restrict__ src, _Float16* __restrict__ dst, int n) {
    int i = blockIdx.x * blockDim.x + threadIdx.x;
    if (i < n) dst[i] = (_Float16)src[i];
}

// ---------------- pack weights into WMMA B-fragment layout ----------------
// src: [K][128] f32 row-major.  frag f = nt*numKt + kt; within frag: lane l
// holds n = nt*16 + (l&15), 16 halfs j -> K = kt*32 + (l>=16?8:0) + (j<8 ? j : j+8).
__global__ void k_pack_w(const float* __restrict__ src, _Float16* __restrict__ dst,
                         int numKt, int total) {
    int t = blockIdx.x * blockDim.x + threadIdx.x;
    if (t >= total) return;
    int frag = t >> 9;
    int within = t & 511;
    int lane = within >> 4;
    int j = within & 15;
    int nt = frag / numKt;
    int kt = frag - nt * numKt;
    int k = kt * 32 + ((lane >= 16) ? 8 : 0) + ((j < 8) ? j : j + 8);
    int n = nt * 16 + (lane & 15);
    dst[t] = (_Float16)src[k * HH + n];
}

// ---------------- edge MLP + attention + scatter ----------------
__global__ void __launch_bounds__(256)
k_edge(const _Float16* __restrict__ h16,
       const float* __restrict__ coord,
       const int* __restrict__ eidx,      // [2][E]
       const _Float16* __restrict__ W1p,
       const float* __restrict__ b1,
       const _Float16* __restrict__ W2p,
       const float* __restrict__ b2,
       const float* __restrict__ W1last,  // W1 row 256, 128 f32
       const float* __restrict__ Wa,
       const float* __restrict__ ba,
       float* __restrict__ agg,
       int E, int num_tiles) {
    __shared__ float4 ein[16 * 33];   // 16 rows x 264 halfs (256 + 8 pad)
    __shared__ float4 m1[16 * 17];    // 16 rows x 136 halfs (128 + 8 pad)
    __shared__ float  m2[16 * 132];   // 16 rows x 132 f32   (128 + 4 pad)
    __shared__ float  radial[16];
    __shared__ int    rows[16];
    __shared__ int    cols[16];
    __shared__ float  att_acc[16];
    __shared__ float  att[16];

    const int t      = threadIdx.x;
    const int lane   = t & 31;
    const int w      = t >> 5;        // wave id = output N-tile (0..7)
    const int nlocal = lane & 15;
    const int hi     = lane >> 4;     // 0: lanes 0-15, 1: lanes 16-31

    // preload all B fragments into registers (reused across all tiles)
    v16h B1[8], B2[4];
    const v16h* W1v = (const v16h*)W1p;   // 32 v16h per fragment
    const v16h* W2v = (const v16h*)W2p;
#pragma unroll
    for (int kt = 0; kt < 8; ++kt) B1[kt] = W1v[(w * 8 + kt) * 32 + lane];
#pragma unroll
    for (int kt = 0; kt < 4; ++kt) B2[kt] = W2v[(w * 4 + kt) * 32 + lane];

    const float bias1 = b1[w * 16 + nlocal];
    const float bias2 = b2[w * 16 + nlocal];
    const float w1l   = W1last[w * 16 + nlocal];
    const float ba0   = ba[0];

    // per-thread staging geometry (constant across tiles)
    const int e_st  = t >> 4;         // edge row this thread stages
    const int seg   = t & 15;         // 32B segment within the 512B row
    const unsigned ein_off = (unsigned)(uintptr_t)&ein[e_st * 33 + seg * 2];

    for (int tile = blockIdx.x; tile < num_tiles; tile += gridDim.x) {
        const int e0 = tile * 16;
        if (t < 16) {
            int r = eidx[e0 + t];
            int c = eidx[E + e0 + t];
            rows[t] = r;
            cols[t] = c;
            float dx = coord[3 * r + 0] - coord[3 * c + 0];
            float dy = coord[3 * r + 1] - coord[3 * c + 1];
            float dz = coord[3 * r + 2] - coord[3 * c + 2];
            radial[t] = dx * dx + dy * dy + dz * dz;
            att_acc[t] = ba0;
            // prefetch next tile's edge indices (global_prefetch_b8)
            int tnext = tile + gridDim.x;
            if (tnext < num_tiles) {
                __builtin_prefetch(&eidx[tnext * 16 + t], 0, 0);
                __builtin_prefetch(&eidx[E + tnext * 16 + t], 0, 0);
            }
        }
        __syncthreads();
        // stage edge-input tile [16][h_row(128) | h_col(128)] as f16:
        // async global->LDS (bypasses VGPRs, ASYNCcnt-tracked)
        {
            int node = (seg < 8) ? rows[e_st] : cols[e_st];
            const _Float16* src = h16 + (size_t)node * 128 + (seg & 7) * 16;
            async_copy_b128(ein_off, src);
            async_copy_b128(ein_off + 16, (const char*)src + 16);
        }
        wait_asynccnt0();
        __syncthreads();

        // ---- layer 1: [16,256] @ W1 + (b1 + radial * W1[256]) ----
        v8f c;
#pragma unroll
        for (int v = 0; v < 8; ++v) c[v] = bias1 + radial[v + hi * 8] * w1l;
        {
            FragU a_cur, a_nxt;
            int f4base = nlocal * 33 + hi;
            a_cur.f4[0] = ein[f4base];
            a_cur.f4[1] = ein[f4base + 2];
#pragma unroll
            for (int kt = 0; kt < 8; ++kt) {
                if (kt < 7) {
                    int f4i = f4base + (kt + 1) * 4;
                    a_nxt.f4[0] = ein[f4i];
                    a_nxt.f4[1] = ein[f4i + 2];
                }
                c = __builtin_amdgcn_wmma_f32_16x16x32_f16(false, a_cur.h, false, B1[kt],
                                                           (short)0, c, false, false);
                a_cur = a_nxt;
            }
        }
        // SiLU -> m1 (f16)
        {
            _Float16* m1h = (_Float16*)m1;
#pragma unroll
            for (int v = 0; v < 8; ++v)
                m1h[(v + hi * 8) * 136 + w * 16 + nlocal] = (_Float16)silu_f(c[v]);
        }
        __syncthreads();

        // ---- layer 2: [16,128] @ W2 + b2 ----
#pragma unroll
        for (int v = 0; v < 8; ++v) c[v] = bias2;
        {
            FragU a_cur, a_nxt;
            int f4base = nlocal * 17 + hi;
            a_cur.f4[0] = m1[f4base];
            a_cur.f4[1] = m1[f4base + 2];
#pragma unroll
            for (int kt = 0; kt < 4; ++kt) {
                if (kt < 3) {
                    int f4i = f4base + (kt + 1) * 4;
                    a_nxt.f4[0] = m1[f4i];
                    a_nxt.f4[1] = m1[f4i + 2];
                }
                c = __builtin_amdgcn_wmma_f32_16x16x32_f16(false, a_cur.h, false, B2[kt],
                                                           (short)0, c, false, false);
                a_cur = a_nxt;
            }
        }
        // SiLU -> m2 (f32)
#pragma unroll
        for (int v = 0; v < 8; ++v)
            m2[(v + hi * 8) * 132 + w * 16 + nlocal] = silu_f(c[v]);
        __syncthreads();

        // ---- attention gate: parallel partial dots + LDS atomic reduce ----
        if (t < 128) {
            int e = t >> 3, ch = t & 7;
            int base = ch * 16;
            float s = 0.f;
#pragma unroll
            for (int j = 0; j < 16; ++j) s += m2[e * 132 + base + j] * Wa[base + j];
            atomicAdd(&att_acc[e], s);          // ds_add_f32
        }
        __syncthreads();
        if (t < 16) att[t] = sigmoid_f(att_acc[t]);
        __syncthreads();

        // ---- scatter-add to agg[row] ----
#pragma unroll
        for (int rep = 0; rep < 8; ++rep) {
            int idx = rep * 256 + t;
            int e = idx >> 7, n = idx & 127;
            unsafeAtomicAdd(&agg[(size_t)rows[e] * 128 + n], m2[e * 132 + n] * att[e]);
        }
        __syncthreads();
    }
}

// ---------------- node MLP + residual ----------------
__global__ void __launch_bounds__(256)
k_node(const _Float16* __restrict__ h16,
       const float* __restrict__ h,
       const float* __restrict__ agg,
       const _Float16* __restrict__ W3p,
       const float* __restrict__ b3,
       const _Float16* __restrict__ W4p,
       const float* __restrict__ b4,
       float* __restrict__ out,
       int num_tiles) {
    __shared__ float4 nin[16 * 33];   // [16][h(128) | agg(128)] f16, stride 264
    __shared__ float4 m3[16 * 17];    // [16][128] f16, stride 136

    const int t      = threadIdx.x;
    const int lane   = t & 31;
    const int w      = t >> 5;
    const int nlocal = lane & 15;
    const int hi     = lane >> 4;

    v16h B3[8], B4[4];
    const v16h* W3v = (const v16h*)W3p;
    const v16h* W4v = (const v16h*)W4p;
#pragma unroll
    for (int kt = 0; kt < 8; ++kt) B3[kt] = W3v[(w * 8 + kt) * 32 + lane];
#pragma unroll
    for (int kt = 0; kt < 4; ++kt) B4[kt] = W4v[(w * 4 + kt) * 32 + lane];

    const float bias3 = b3[w * 16 + nlocal];
    const float bias4 = b4[w * 16 + nlocal];

    const int e_st = t >> 4;
    const int seg  = t & 15;
    const unsigned nin_off = (unsigned)(uintptr_t)&nin[e_st * 33 + seg * 2];

    for (int tile = blockIdx.x; tile < num_tiles; tile += gridDim.x) {
        // stage node-input tile [16][h(128,f16 async) | agg(128,f32->f16)]
        {
            int node = tile * 16 + e_st;
            if (seg < 8) {
                const _Float16* src = h16 + (size_t)node * 128 + seg * 16;
                async_copy_b128(nin_off, src);
                async_copy_b128(nin_off + 16, (const char*)src + 16);
            } else {
                const float* src = agg + (size_t)node * 128 + (seg - 8) * 16;
                _Float16* row = (_Float16*)&nin[e_st * 33];
#pragma unroll
                for (int j = 0; j < 16; ++j)
                    row[128 + (seg - 8) * 16 + j] = (_Float16)src[j];
            }
        }
        wait_asynccnt0();
        __syncthreads();

        // ---- layer 3: [16,256] @ W3 + b3 ----
        v8f c;
#pragma unroll
        for (int v = 0; v < 8; ++v) c[v] = bias3;
        {
            FragU a_cur, a_nxt;
            int f4base = nlocal * 33 + hi;
            a_cur.f4[0] = nin[f4base];
            a_cur.f4[1] = nin[f4base + 2];
#pragma unroll
            for (int kt = 0; kt < 8; ++kt) {
                if (kt < 7) {
                    int f4i = f4base + (kt + 1) * 4;
                    a_nxt.f4[0] = nin[f4i];
                    a_nxt.f4[1] = nin[f4i + 2];
                }
                c = __builtin_amdgcn_wmma_f32_16x16x32_f16(false, a_cur.h, false, B3[kt],
                                                           (short)0, c, false, false);
                a_cur = a_nxt;
            }
        }
        {
            _Float16* m3h = (_Float16*)m3;
#pragma unroll
            for (int v = 0; v < 8; ++v)
                m3h[(v + hi * 8) * 136 + w * 16 + nlocal] = (_Float16)silu_f(c[v]);
        }
        __syncthreads();

        // ---- layer 4: [16,128] @ W4 + b4, residual, store ----
#pragma unroll
        for (int v = 0; v < 8; ++v) c[v] = bias4;
        {
            FragU a_cur, a_nxt;
            int f4base = nlocal * 17 + hi;
            a_cur.f4[0] = m3[f4base];
            a_cur.f4[1] = m3[f4base + 2];
#pragma unroll
            for (int kt = 0; kt < 4; ++kt) {
                if (kt < 3) {
                    int f4i = f4base + (kt + 1) * 4;
                    a_nxt.f4[0] = m3[f4i];
                    a_nxt.f4[1] = m3[f4i + 2];
                }
                c = __builtin_amdgcn_wmma_f32_16x16x32_f16(false, a_cur.h, false, B4[kt],
                                                           (short)0, c, false, false);
                a_cur = a_nxt;
            }
        }
        const int n = w * 16 + nlocal;
#pragma unroll
        for (int v = 0; v < 8; ++v) {
            size_t node = (size_t)(tile * 16 + v + hi * 8);
            out[node * 128 + n] = h[node * 128 + n] + c[v];
        }
        __syncthreads();
    }
}

extern "C" void kernel_launch(void* const* d_in, const int* in_sizes, int n_in,
                              void* d_out, int out_size, void* d_ws, size_t ws_size,
                              hipStream_t stream) {
    const float* h     = (const float*)d_in[0];
    const float* coord = (const float*)d_in[1];
    const int*   eidx  = (const int*)d_in[2];
    const float* W1    = (const float*)d_in[3];
    const float* b1    = (const float*)d_in[4];
    const float* W2    = (const float*)d_in[5];
    const float* b2    = (const float*)d_in[6];
    const float* Wa    = (const float*)d_in[7];
    const float* ba    = (const float*)d_in[8];
    const float* W3    = (const float*)d_in[9];
    const float* b3    = (const float*)d_in[10];
    const float* W4    = (const float*)d_in[11];
    const float* b4    = (const float*)d_in[12];
    float* out = (float*)d_out;

    char* ws = (char*)d_ws;
    _Float16* h16 = (_Float16*)(ws + OFF_H16);
    _Float16* W1p = (_Float16*)(ws + OFF_W1P);
    _Float16* W2p = (_Float16*)(ws + OFF_W2P);
    _Float16* W3p = (_Float16*)(ws + OFF_W3P);
    _Float16* W4p = (_Float16*)(ws + OFF_W4P);
    float*    agg = (float*)(ws + OFF_AGG);

    const int N = in_sizes[0] / DD;   // 40000
    const int E = in_sizes[2] / 2;    // 640000
    const int nh = N * DD;            // 5,120,000

    // zero agg (workspace is not re-poisoned between timed replays)
    k_zero_f4<<<(nh / 4 + 255) / 256, 256, 0, stream>>>((float4*)agg, nh / 4);
    // convert h to f16
    k_cvt_h16<<<(nh + 255) / 256, 256, 0, stream>>>(h, h16, nh);
    // pack weights into WMMA B-fragment layout
    k_pack_w<<<(8 * 8 * 512) / 256, 256, 0, stream>>>(W1, W1p, 8, 8 * 8 * 512);
    k_pack_w<<<(8 * 4 * 512) / 256, 256, 0, stream>>>(W2, W2p, 4, 8 * 4 * 512);
    k_pack_w<<<(8 * 8 * 512) / 256, 256, 0, stream>>>(W3, W3p, 8, 8 * 8 * 512);
    k_pack_w<<<(8 * 4 * 512) / 256, 256, 0, stream>>>(W4, W4p, 4, 8 * 4 * 512);

    // edge MLP + attention + scatter
    k_edge<<<2048, 256, 0, stream>>>(h16, coord, eidx, W1p, b1, W2p, b2,
                                     W1 + 256 * HH, Wa, ba, agg, E, E / 16);
    // node MLP + residual
    k_node<<<1250, 256, 0, stream>>>(h16, h, agg, W3p, b3, W4p, b4, out, N / 16);

    // coord passthrough (second tuple element)
    hipMemcpyAsync(out + (size_t)N * DD, coord, (size_t)N * 3 * sizeof(float),
                   hipMemcpyDeviceToDevice, stream);
}
